// GCN_59416577573473
// MI455X (gfx1250) — matrix operand order repaired and checked by reading, not compiled, not run on hardware
//
#include <hip/hip_runtime.h>
#include <hip/hip_bf16.h>
#include <math.h>

typedef __attribute__((ext_vector_type(16))) __bf16 v16bf;
typedef __attribute__((ext_vector_type(8)))  float  v8f;
typedef __attribute__((ext_vector_type(4)))  float  vf4;

union FragAB { v16bf v; vf4 q[2]; };

__device__ __forceinline__ v16bf load_frag2(const __bf16* p0, const __bf16* p1) {
  FragAB f;
  f.q[0] = *(const vf4*)p0;
  f.q[1] = *(const vf4*)p1;
  return f.v;
}

#define WMMA_BF16(A_, B_, C_) \
  __builtin_amdgcn_wmma_f32_16x16x32_bf16(false, (A_), false, (B_), (short)0, (C_), false, false)

// ---------------------------------------------------------------------------
// WMMA GEMM: C[M,Nout] (f32) = A[M,K] (bf16, row-major) * Bt[Nout,K]^T (bf16)
// One wave computes a 32x64 tile via 2x4 v_wmma_f32_16x16x32_bf16
// (A fragments reused across 4 B fragments -> 12 16B loads per 8 WMMA).
// A-fragment (16x32, lane m=lane&15, half=lane>>4): K chunk
// [half*16,+16) contiguous of row m. B-fragment (32x16): column n=lane&15, K chunk
// [half*16,+16) contiguous from the transposed weight row n.
// ---------------------------------------------------------------------------
__global__ void __launch_bounds__(256)
gemm_bf16_wmma(const __bf16* __restrict__ A,
               const __bf16* __restrict__ Bt,
               float* __restrict__ C,
               int M, int Nout, int K) {
  const int lane = threadIdx.x & 31;
  const int wid  = (int)((blockIdx.x * blockDim.x + threadIdx.x) >> 5);
  const int ntiles = Nout >> 6;            // 64-wide column tiles
  const int tm = (wid / ntiles) << 5;      // 32 rows per wave
  const int tn = (wid % ntiles) << 6;      // 64 cols per wave
  if (tm >= M) return;  // wave-uniform: EXEC stays all-ones for active waves

  const int half = lane >> 4;
  const int l15  = lane & 15;

  const __bf16* a0 = A + (size_t)(tm + l15) * K;
  const __bf16* a1 = a0 + (size_t)16 * K;
  const __bf16* b0 = Bt + (size_t)(tn + l15) * K;
  const __bf16* b1 = b0 + (size_t)16 * K;
  const __bf16* b2 = b0 + (size_t)32 * K;
  const __bf16* b3 = b0 + (size_t)48 * K;

  v8f c00 = {}, c01 = {}, c02 = {}, c03 = {};
  v8f c10 = {}, c11 = {}, c12 = {}, c13 = {};

  for (int k = 0; k < K; k += 32) {
    if (k + 32 < K) {
      __builtin_prefetch(a0 + k + 32, 0, 1);
      __builtin_prefetch(a1 + k + 32, 0, 1);
    }
    v16bf fa0 = load_frag2(a0 + k + half * 16, a0 + k + half * 16 + 8);
    v16bf fa1 = load_frag2(a1 + k + half * 16, a1 + k + half * 16 + 8);
    v16bf fb0 = load_frag2(b0 + k + half * 16, b0 + k + half * 16 + 8);
    v16bf fb1 = load_frag2(b1 + k + half * 16, b1 + k + half * 16 + 8);
    v16bf fb2 = load_frag2(b2 + k + half * 16, b2 + k + half * 16 + 8);
    v16bf fb3 = load_frag2(b3 + k + half * 16, b3 + k + half * 16 + 8);
    c00 = WMMA_BF16(fa0, fb0, c00);
    c01 = WMMA_BF16(fa0, fb1, c01);
    c02 = WMMA_BF16(fa0, fb2, c02);
    c03 = WMMA_BF16(fa0, fb3, c03);
    c10 = WMMA_BF16(fa1, fb0, c10);
    c11 = WMMA_BF16(fa1, fb1, c11);
    c12 = WMMA_BF16(fa1, fb2, c12);
    c13 = WMMA_BF16(fa1, fb3, c13);
  }

  // C/D layout: lane 0-15 -> row tm+v, lane 16-31 -> row tm+8+v
  const int rb = tm + half * 8;
  float* cr = C + (size_t)rb * Nout + tn + l15;
#pragma unroll
  for (int v = 0; v < 8; ++v) {
    cr[(size_t)v * Nout +  0] = c00[v];
    cr[(size_t)v * Nout + 16] = c01[v];
    cr[(size_t)v * Nout + 32] = c02[v];
    cr[(size_t)v * Nout + 48] = c03[v];
    cr[(size_t)(v + 16) * Nout +  0] = c10[v];
    cr[(size_t)(v + 16) * Nout + 16] = c11[v];
    cr[(size_t)(v + 16) * Nout + 32] = c12[v];
    cr[(size_t)(v + 16) * Nout + 48] = c13[v];
  }
}

// ---------------------------------------------------------------------------
// Graph plumbing kernels (shift/mask indexing; wt_cvt and sigmoid_out use integer division)
// ---------------------------------------------------------------------------
__global__ void deg_scatter(const int* __restrict__ dst, int E, float* __restrict__ deg) {
  int i = blockIdx.x * blockDim.x + threadIdx.x;
  if (i < E) atomicAdd(&deg[dst[i]], 1.0f);
}

__global__ void deg_finalize(const float* __restrict__ deg, float* __restrict__ dis,
                             float* __restrict__ invd, int N) {
  int i = blockIdx.x * blockDim.x + threadIdx.x;
  if (i < N) {
    float d = deg[i] + 1.0f;   // +1 self-loop
    dis[i]  = rsqrtf(d);
    invd[i] = 1.0f / d;
  }
}

// Wt[fo, k] = W[k, fo]   (W is [K, Fout] row-major); pad rows fo >= Fout with 0
__global__ void wt_cvt(const float* __restrict__ W, __bf16* __restrict__ Wt,
                       int K, int Fout, int FoutPad) {
  int i = blockIdx.x * blockDim.x + threadIdx.x;
  if (i < K * FoutPad) {
    int fo = i / K, k = i % K;
    Wt[(size_t)fo * K + k] = (fo < Fout) ? (__bf16)W[(size_t)k * Fout + fo] : (__bf16)0.0f;
  }
}

__global__ void cvt_f32_bf16(const float* __restrict__ in, __bf16* __restrict__ out, size_t n) {
  size_t stride = (size_t)gridDim.x * blockDim.x;
  for (size_t i = (size_t)blockIdx.x * blockDim.x + threadIdx.x; i < n; i += stride)
    out[i] = (__bf16)in[i];
}

// agg[i,:] = h[i,:] * invdeg[i]  (self-loop term; also initializes agg)
// F = 4 << gvSh
__global__ void selfloop_init(const float* __restrict__ h, const float* __restrict__ invd,
                              float* __restrict__ agg, int N, int gvSh) {
  const int F = 4 << gvSh;
  const size_t gmask = (size_t)((1 << gvSh) - 1);
  const size_t total = (size_t)N << gvSh;
  size_t stride = (size_t)gridDim.x * blockDim.x;
  for (size_t i = (size_t)blockIdx.x * blockDim.x + threadIdx.x; i < total; i += stride) {
    size_t node = i >> gvSh, g = i & gmask;
    vf4 hv = *(const vf4*)(h + node * F + 4 * g);
    float s = invd[node];
    *(vf4*)(agg + node * F + 4 * g) = hv * s;
  }
}

// agg[dst,:] += h[src,:] * dis[src] * dis[dst]   (vectorized f32 atomics)
__global__ void edge_scatter(const float* __restrict__ h, const int* __restrict__ src,
                             const int* __restrict__ dst, const float* __restrict__ dis,
                             float* __restrict__ agg, int E, int gvSh) {
  const int F = 4 << gvSh;
  const size_t gmask = (size_t)((1 << gvSh) - 1);
  const size_t total = (size_t)E << gvSh;
  size_t stride = (size_t)gridDim.x * blockDim.x;
  for (size_t i = (size_t)blockIdx.x * blockDim.x + threadIdx.x; i < total; i += stride) {
    size_t e = i >> gvSh, g = i & gmask;
    int s = src[e], d = dst[e];
    float nrm = dis[s] * dis[d];
    vf4 hv = *(const vf4*)(h + (size_t)s * F + 4 * g);
    float* ap = agg + (size_t)d * F + 4 * g;
    atomicAdd(ap + 0, hv.x * nrm);
    atomicAdd(ap + 1, hv.y * nrm);
    atomicAdd(ap + 2, hv.z * nrm);
    atomicAdd(ap + 3, hv.w * nrm);
  }
}

// act = tanh(in + b) as bf16; optionally also write f32 copy. F power of two.
__global__ void bias_act(const float* __restrict__ in, const float* __restrict__ b,
                         __bf16* __restrict__ actout, float* __restrict__ f32out,
                         int N, int F) {
  const size_t total = (size_t)N * F;
  const size_t fmask = (size_t)(F - 1);
  size_t stride = (size_t)gridDim.x * blockDim.x;
  for (size_t i = (size_t)blockIdx.x * blockDim.x + threadIdx.x; i < total; i += stride) {
    int f = (int)(i & fmask);
    float v = tanhf(in[i] + b[f]);
    actout[i] = (__bf16)v;
    if (f32out) f32out[i] = v;
  }
}

// out[N,Fo] = sigmoid(hc[:, :Fo] + bc), hc has padded width Fp
__global__ void sigmoid_out(const float* __restrict__ hc, const float* __restrict__ bc,
                            float* __restrict__ out, int N, int Fp, int Fo) {
  const size_t total = (size_t)N * Fo;
  size_t stride = (size_t)gridDim.x * blockDim.x;
  for (size_t i = (size_t)blockIdx.x * blockDim.x + threadIdx.x; i < total; i += stride) {
    size_t node = i / Fo, f = i % Fo;
    float v = hc[node * Fp + f] + bc[f];
    out[i] = 1.0f / (1.0f + expf(-v));
  }
}

// ---------------------------------------------------------------------------
extern "C" void kernel_launch(void* const* d_in, const int* in_sizes, int n_in,
                              void* d_out, int out_size, void* d_ws, size_t ws_size,
                              hipStream_t stream) {
  const float* x  = (const float*)d_in[0];
  const int*   ei = (const int*)  d_in[1];
  const float* W1 = (const float*)d_in[2];
  const float* b1 = (const float*)d_in[3];
  const float* W2 = (const float*)d_in[4];
  const float* b2 = (const float*)d_in[5];
  const float* W3 = (const float*)d_in[6];
  const float* b3 = (const float*)d_in[7];
  const float* Wc = (const float*)d_in[8];
  const float* bc = (const float*)d_in[9];

  const int N = in_sizes[0] / 256;   // 100000 (divisible by 32)
  const int E = in_sizes[1] / 2;     // 400000
  const int* src  = ei;
  const int* dstv = ei + E;

  float* out = (float*)d_out;              // [N,100]
  float* h3o = out + (size_t)N * 100;      // [N,128]

  // ---- workspace carve (256B-aligned) ----
  char* wsp = (char*)d_ws;
  auto carve = [&](size_t bytes) -> char* {
    char* r = wsp;
    wsp += (bytes + 255) & ~(size_t)255;
    return r;
  };
  float*  deg  = (float*) carve((size_t)N * 4);
  float*  dis  = (float*) carve((size_t)N * 4);
  float*  invd = (float*) carve((size_t)N * 4);
  __bf16* W1t  = (__bf16*)carve((size_t)1024 * 256 * 2);
  __bf16* W2t  = (__bf16*)carve((size_t)512 * 1024 * 2);
  __bf16* W3t  = (__bf16*)carve((size_t)128 * 512 * 2);
  __bf16* Wct  = (__bf16*)carve((size_t)128 * 128 * 2);
  __bf16* act  = (__bf16*)carve((size_t)N * 1024 * 2);
  float*  hbuf = (float*) carve((size_t)N * 1024 * 4);
  float*  aggb = (float*) carve((size_t)N * 1024 * 4);

  // ---- degree / normalization ----
  hipMemsetAsync(deg, 0, (size_t)N * 4, stream);
  deg_scatter <<<(E + 255) / 256, 256, 0, stream>>>(dstv, E, deg);
  deg_finalize<<<(N + 255) / 256, 256, 0, stream>>>(deg, dis, invd, N);

  // ---- weights -> transposed bf16 ----
  wt_cvt<<<(256 * 1024 + 255) / 256, 256, 0, stream>>>(W1, W1t, 256, 1024, 1024);
  wt_cvt<<<(1024 * 512 + 255) / 256, 256, 0, stream>>>(W2, W2t, 1024, 512, 512);
  wt_cvt<<<(512 * 128 + 255) / 256, 256, 0, stream>>>(W3, W3t, 512, 128, 128);
  wt_cvt<<<(128 * 128 + 255) / 256, 256, 0, stream>>>(Wc, Wct, 128, 100, 128);

  // ---- layer 1: aggregate FIRST on 256-dim input (agg(x)@W == agg(x@W)),
  //      4x less atomic traffic than scattering the 1024-dim GEMM output ----
  {
    const int K = 256, F = 1024;
    selfloop_init<<<4096, 256, 0, stream>>>(x, invd, aggb, N, 6);       // F=256
    edge_scatter <<<16384, 256, 0, stream>>>(x, src, dstv, dis, aggb, E, 6);
    cvt_f32_bf16 <<<4096, 256, 0, stream>>>(aggb, act, (size_t)N * 256);
    const int waves = (N / 32) * (F / 64);
    gemm_bf16_wmma<<<(waves + 7) / 8, 256, 0, stream>>>(act, W1t, hbuf, N, F, K);
    bias_act<<<8192, 256, 0, stream>>>(hbuf, b1, act, nullptr, N, F);
  }
  // ---- layer 2: 1024 -> 512, aggregate after GEMM (output side smaller) ----
  {
    const int K = 1024, F = 512;
    const int waves = (N / 32) * (F / 64);
    gemm_bf16_wmma<<<(waves + 7) / 8, 256, 0, stream>>>(act, W2t, hbuf, N, F, K);
    selfloop_init<<<4096, 256, 0, stream>>>(hbuf, invd, aggb, N, 7);    // F=512
    edge_scatter <<<16384, 256, 0, stream>>>(hbuf, src, dstv, dis, aggb, E, 7);
    bias_act     <<<8192, 256, 0, stream>>>(aggb, b2, act, nullptr, N, F);
  }
  // ---- layer 3: 512 -> 128, aggregate after GEMM (h output) ----
  {
    const int K = 512, F = 128;
    const int waves = (N / 32) * (F / 64);
    gemm_bf16_wmma<<<(waves + 7) / 8, 256, 0, stream>>>(act, W3t, hbuf, N, F, K);
    selfloop_init<<<4096, 256, 0, stream>>>(hbuf, invd, aggb, N, 5);    // F=128
    edge_scatter <<<16384, 256, 0, stream>>>(hbuf, src, dstv, dis, aggb, E, 5);
    bias_act     <<<8192, 256, 0, stream>>>(aggb, b3, act, h3o, N, F);
  }
  // ---- classifier: 128 -> 100 (padded to 128) ----
  {
    const int K = 128, F = 128;
    const int waves = (N / 32) * (F / 64);
    gemm_bf16_wmma<<<(waves + 7) / 8, 256, 0, stream>>>(act, Wct, hbuf, N, F, K);
    sigmoid_out<<<8192, 256, 0, stream>>>(hbuf, bc, out, N, 128, 100);
  }
}
